// minerva_10969346474458
// MI455X (gfx1250) — matrix operand order repaired and checked by reading, not compiled, not run on hardware
//
#include <hip/hip_runtime.h>
#include <hip/hip_bf16.h>
#include <math.h>

// ---------------- problem constants ----------------
#define BB    1024
#define DD    768
#define NN    50000
#define LL    28
#define NPAD  50016          // 32 * 1563
#define NCHUNK 1563          // NPAD / 32
#define SLICES 16            // grid.y for echo kernel
#define NSLOT  (SLICES * 8)  // 128 wave slots per row-tile
#define SCALE_A 4096.0f      // keeps s^3 out of f16 subnormal range

typedef _Float16 v16h __attribute__((ext_vector_type(16)));
typedef _Float16 v8h  __attribute__((ext_vector_type(8)));
typedef float    v8f  __attribute__((ext_vector_type(8)));
typedef float    v4f  __attribute__((ext_vector_type(4)));

static __device__ __forceinline__ v8f wmma_f16(v16h a, v16h b, v8f c) {
    return __builtin_amdgcn_wmma_f32_16x16x32_f16(false, a, false, b, (short)0, c,
                                                  false, false);
}

// ---------------- K1: convert W_g (f32) -> f16 ----------------
__global__ __launch_bounds__(256) void cvt_w_kernel(const float* __restrict__ src,
                                                    _Float16* __restrict__ dst, int n) {
    int i = blockIdx.x * 256 + threadIdx.x;
    if (i < n) dst[i] = (_Float16)src[i];
}

// ---------------- K2: ex_classes [N,28] f32 -> clsT [32,NPAD] f16 (zero padded) ----
__global__ __launch_bounds__(256) void cls_fill_kernel(const float* __restrict__ exc,
                                                       _Float16* __restrict__ clsT) {
    int idx = blockIdx.x * 256 + threadIdx.x;
    if (idx >= 32 * NPAD) return;
    int l = idx / NPAD;
    int n = idx - l * NPAD;
    float v = (l < LL && n < NN) ? exc[(size_t)n * LL + l] : 0.0f;
    clsT[idx] = (_Float16)v;
}

// ---------------- K5: per-class exemplar counts ----------------
__global__ __launch_bounds__(256) void counts_kernel(const float* __restrict__ exc,
                                                     float* __restrict__ counts) {
    __shared__ float red[256];
    const int l = blockIdx.x;                  // 0..27
    float s = 0.0f;
    for (int n = threadIdx.x; n < NN; n += 256) s += exc[(size_t)n * LL + l];
    red[threadIdx.x] = s;
    __syncthreads();
    for (int st = 128; st > 0; st >>= 1) {
        if (threadIdx.x < st) red[threadIdx.x] += red[threadIdx.x + st];
        __syncthreads();
    }
    if (threadIdx.x == 0) counts[l] = red[0];
}

// ---------------- K3: Y = l2norm_rows(X @ W^T) as f16 -----------------------------
// One workgroup (8 waves) computes 16 rows x 768 cols. Wave w owns 6 col-tiles.
// WMMA A: lane m holds row m, K = hi*16 + [0..15]  (contiguous in X row).
// WMMA B: lane holds col = tile*16 + lm, K = hi*16 + [0..15]  (contiguous in Wh row).
__global__ __launch_bounds__(256) void proj_norm_kernel(const float* __restrict__ X,
                                                        const _Float16* __restrict__ Wh,
                                                        _Float16* __restrict__ Yh,
                                                        int nrows) {
    __shared__ float Y[16][DD + 4];
    __shared__ float part[16][16];
    __shared__ float invn[16];
    const int tid  = threadIdx.x;
    const int wv   = tid >> 5;
    const int lane = tid & 31;
    const int lm   = lane & 15;
    const int hi   = lane >> 4;
    const int row0 = blockIdx.x * 16;
    const int r    = row0 + lm;
    const bool rok = r < nrows;

    v8f c[6];
    for (int t = 0; t < 6; ++t) c[t] = (v8f)0.0f;

    for (int kt = 0; kt < DD / 32; ++kt) {
        const int k0 = kt * 32;
        // A fragment: load fp32, convert to f16 in-register
        v4f x0 = (v4f)0.f, x1 = (v4f)0.f, x2 = (v4f)0.f, x3 = (v4f)0.f;
        if (rok) {
            const float* p = X + (size_t)r * DD + k0 + hi * 16;
            x0 = *(const v4f*)(p);
            x1 = *(const v4f*)(p + 4);
            x2 = *(const v4f*)(p + 8);
            x3 = *(const v4f*)(p + 12);
        }
        v16h a;
        for (int j = 0; j < 4; ++j) {
            a[j]      = (_Float16)x0[j];
            a[4 + j]  = (_Float16)x1[j];
            a[8 + j]  = (_Float16)x2[j];
            a[12 + j] = (_Float16)x3[j];
        }
        for (int t = 0; t < 6; ++t) {
            const int col = (wv * 6 + t) * 16 + lm;
            v16h b = *(const v16h*)(Wh + (size_t)col * DD + k0 + hi * 16);
            c[t] = wmma_f16(a, b, c[t]);
        }
    }
    // C fragment -> LDS tile (M = v + 8*hi, N = col)
    for (int t = 0; t < 6; ++t) {
        const int col = (wv * 6 + t) * 16 + lm;
        for (int v = 0; v < 8; ++v) Y[v + 8 * hi][col] = c[t][v];
    }
    __syncthreads();
    // row L2 norms
    {
        const int rr = tid >> 4, jj = tid & 15;
        float s = 0.0f;
        for (int i = 0; i < DD / 16; ++i) {
            float y = Y[rr][jj + 16 * i];
            s += y * y;
        }
        part[rr][jj] = s;
    }
    __syncthreads();
    if (tid < 16) {
        float s = 0.0f;
        for (int j = 0; j < 16; ++j) s += part[tid][j];
        invn[tid] = 1.0f / fmaxf(sqrtf(s), 1e-12f);
    }
    __syncthreads();
    for (int idx = tid; idx < 16 * DD; idx += 256) {
        int rr = idx / DD, cc = idx - rr * DD;
        Yh[(size_t)(row0 + rr) * DD + cc] = (_Float16)(Y[rr][cc] * invn[rr]);
    }
}

// ---------------- K4: fused  s = Fn @ En^T ; a = (s^3)*SCALE ; echoP += a @ clsT^T --
// grid = (64 row-tiles, 16 slices), block = 256 (8 waves).
// Each wave owns chunks of 32 exemplars strided by 128 slots, accumulates a private
// 16x32 echo partial, writes it to ws (deterministic reduction later).
__global__ __launch_bounds__(256) void echo_kernel(const _Float16* __restrict__ Fh,
                                                   const _Float16* __restrict__ Eh,
                                                   const _Float16* __restrict__ clsT,
                                                   float* __restrict__ epart) {
    __shared__ alignas(32) _Float16 lt[8][16][32];   // per-wave transpose buffer
    const int tid  = threadIdx.x;
    const int wv   = tid >> 5;
    const int lane = tid & 31;
    const int lm   = lane & 15;
    const int hi   = lane >> 4;
    const int rt   = blockIdx.x;                 // row tile 0..63
    const int slot = blockIdx.y * 8 + wv;        // 0..127

    v8f ec0 = (v8f)0.0f, ec1 = (v8f)0.0f;
    const _Float16* fRow = Fh + (size_t)(rt * 16 + lm) * DD;

    for (int ch = slot; ch < NCHUNK; ch += NSLOT) {
        const int n0 = ch * 32;
        v8f c0 = (v8f)0.0f, c1 = (v8f)0.0f;
        const _Float16* e0p = Eh + (size_t)(n0 + lm) * DD + hi * 16;
        const _Float16* e1p = e0p + (size_t)16 * DD;
        for (int kt = 0; kt < DD / 32; ++kt) {
            const int k0 = kt * 32;
            v8h alo = *(const v8h*)(fRow + k0 + hi * 16);
            v8h ahi = *(const v8h*)(fRow + k0 + hi * 16 + 8);
            v16h a = __builtin_shufflevector(alo, ahi, 0, 1, 2, 3, 4, 5, 6, 7,
                                             8, 9, 10, 11, 12, 13, 14, 15);
            v16h b0 = *(const v16h*)(e0p + k0);
            v16h b1 = *(const v16h*)(e1p + k0);
            c0 = wmma_f16(a, b0, c0);
            c1 = wmma_f16(a, b1, c1);
        }
        // cube + scale, scatter to LDS in (row, n) layout
        for (int v = 0; v < 8; ++v) {
            float s0 = c0[v], s1 = c1[v];
            lt[wv][v + 8 * hi][lm]      = (_Float16)(s0 * s0 * s0 * SCALE_A);
            lt[wv][v + 8 * hi][16 + lm] = (_Float16)(s1 * s1 * s1 * SCALE_A);
        }
        // cross-lane LDS dependency inside one wave: wait on the DS counter
        asm volatile("s_wait_dscnt 0" ::: "memory");
        // gather back as an A fragment (row = lm, K = hi*16 + [0..15])
        v8h t0 = *(const v8h*)&lt[wv][lm][hi * 16];
        v8h t1 = *(const v8h*)&lt[wv][lm][hi * 16 + 8];
        v16h ae = __builtin_shufflevector(t0, t1, 0, 1, 2, 3, 4, 5, 6, 7,
                                          8, 9, 10, 11, 12, 13, 14, 15);
        v16h bc0 = *(const v16h*)(clsT + (size_t)lm * NPAD + n0 + hi * 16);
        v16h bc1 = *(const v16h*)(clsT + (size_t)(16 + lm) * NPAD + n0 + hi * 16);
        ec0 = wmma_f16(ae, bc0, ec0);
        ec1 = wmma_f16(ae, bc1, ec1);
        if (ch + NSLOT < NCHUNK) {
            __builtin_prefetch((const void*)(Eh + (size_t)((ch + NSLOT) * 32 + lm) * DD),
                               0, 1);
        }
    }
    // write this wave's 16x32 partial (deterministic reduction downstream)
    float* base = epart + ((size_t)rt * NSLOT + slot) * 512;
    for (int v = 0; v < 8; ++v) {
        base[(v + 8 * hi) * 32 + lm]      = ec0[v];
        base[(v + 8 * hi) * 32 + 16 + lm] = ec1[v];
    }
}

// ---------------- K6: deterministic reduction of echo partials --------------------
__global__ __launch_bounds__(256) void reduce_echo_kernel(const float* __restrict__ epart,
                                                          float* __restrict__ echo_acc) {
    int idx = blockIdx.x * 256 + threadIdx.x;   // 0..32767  (b*32 + l)
    if (idx >= BB * 32) return;
    int b = idx >> 5, l = idx & 31;
    int rt = b >> 4, ml = b & 15;
    const float* p = epart + (size_t)rt * NSLOT * 512 + ml * 32 + l;
    float s = 0.0f;
    for (int slot = 0; slot < NSLOT; ++slot) s += p[(size_t)slot * 512];
    echo_acc[idx] = s;
}

// ---------------- K7: dists + BCE (single block, deterministic) -------------------
__global__ __launch_bounds__(1024) void finalize_kernel(const float* __restrict__ echo_acc,
                                                        const float* __restrict__ counts,
                                                        const float* __restrict__ labels,
                                                        float* __restrict__ out) {
    __shared__ float red[1024];
    const int b = threadIdx.x;                 // one row per thread, 1024 threads
    float e[LL];
    float ss = 0.0f;
    for (int l = 0; l < LL; ++l) {
        float v = echo_acc[b * 32 + l] / (SCALE_A * counts[l]);
        e[l] = v;
        ss += v * v;
    }
    float lsum = 0.0f;
    for (int l = 0; l < LL; ++l) {
        float nd = -sqrtf(fmaxf(ss - 2.0f * e[l] + 1.0f, 0.0f));
        out[1 + b * LL + l] = nd;
        // softplus(nd) - nd*y ; nd <= 0 so log1p(exp(nd)) is stable
        lsum += log1pf(expf(nd)) - nd * labels[b * LL + l];
    }
    red[b] = lsum;
    __syncthreads();
    for (int st = 512; st > 0; st >>= 1) {
        if (b < st) red[b] += red[b + st];
        __syncthreads();
    }
    if (b == 0) out[0] = red[0] * (1.0f / (float)(BB * LL));
}

// ---------------- workspace layout (bytes, 256B aligned blocks) -------------------
static const size_t O_WH     = 0;                                   // 768*768 f16
static const size_t O_FH     = O_WH + (size_t)DD * DD * 2;          // 1024*768 f16
static const size_t O_EH     = O_FH + (size_t)BB * DD * 2;          // NPAD*768 f16
static const size_t O_CLS    = O_EH + (size_t)NPAD * DD * 2;        // 32*NPAD f16
static const size_t O_COUNTS = O_CLS + (size_t)32 * NPAD * 2;       // 28 f32 (pad 256B)
static const size_t O_ECHO   = O_COUNTS + 256;                      // 1024*32 f32
static const size_t O_EPART  = O_ECHO + (size_t)BB * 32 * 4;        // 64*128*512 f32

extern "C" void kernel_launch(void* const* d_in, const int* in_sizes, int n_in,
                              void* d_out, int out_size, void* d_ws, size_t ws_size,
                              hipStream_t stream) {
    const float* features    = (const float*)d_in[0];   // [1024,768]
    const float* labels      = (const float*)d_in[1];   // [1024,28]
    const float* W_g         = (const float*)d_in[2];   // [768,768]
    const float* ex_features = (const float*)d_in[3];   // [50000,768]
    const float* ex_classes  = (const float*)d_in[4];   // [50000,28]
    // d_in[5] = class_reps (identity) -> folded analytically

    char* ws = (char*)d_ws;
    _Float16* Wh     = (_Float16*)(ws + O_WH);
    _Float16* Fh     = (_Float16*)(ws + O_FH);
    _Float16* Eh     = (_Float16*)(ws + O_EH);
    _Float16* clsT   = (_Float16*)(ws + O_CLS);
    float*    counts = (float*)   (ws + O_COUNTS);
    float*    echoA  = (float*)   (ws + O_ECHO);
    float*    epart  = (float*)   (ws + O_EPART);
    float*    out    = (float*)d_out;

    // prep
    cvt_w_kernel<<<(DD * DD + 255) / 256, 256, 0, stream>>>(W_g, Wh, DD * DD);
    cls_fill_kernel<<<(32 * NPAD + 255) / 256, 256, 0, stream>>>(ex_classes, clsT);
    counts_kernel<<<LL, 256, 0, stream>>>(ex_classes, counts);
    // projection + L2 normalize (WMMA)
    proj_norm_kernel<<<BB / 16, 256, 0, stream>>>(features, Wh, Fh, BB);
    proj_norm_kernel<<<NPAD / 16, 256, 0, stream>>>(ex_features, Wh, Eh, NN);
    // fused cosine-GEMM + cube + class-contraction (WMMA x2)
    echo_kernel<<<dim3(BB / 16, SLICES), 256, 0, stream>>>(Fh, Eh, clsT, epart);
    // deterministic reduction + loss
    reduce_echo_kernel<<<(BB * 32 + 255) / 256, 256, 0, stream>>>(epart, echoA);
    finalize_kernel<<<1, 1024, 0, stream>>>(echoA, counts, labels, out);
}
